// TransformerBlock_41180146434895
// MI455X (gfx1250) — matrix operand order repaired and checked by reading, not compiled, not run on hardware
//
#include <hip/hip_runtime.h>
#include <hip/hip_bf16.h>

// ---------------------------------------------------------------------------
// Types for CDNA5 WMMA (wave32): V_WMMA_F32_16X16X32_BF16
// ---------------------------------------------------------------------------
typedef __attribute__((ext_vector_type(16))) __bf16 bf16x16;
typedef __attribute__((ext_vector_type(8)))  __bf16 bf16x8;
typedef __attribute__((ext_vector_type(4)))  __bf16 bf16x4;
typedef __attribute__((ext_vector_type(8)))  float  f32x8;
typedef __attribute__((ext_vector_type(4)))  unsigned int u32x4;
typedef __attribute__((ext_vector_type(8)))  int i32x8;
typedef __attribute__((ext_vector_type(4)))  int i32x4;

#define DEV __device__ __forceinline__

DEV __bf16 f2bf(float f) {
  unsigned u = __float_as_uint(f);
  u += 0x7FFFu + ((u >> 16) & 1u);           // round-to-nearest-even
  unsigned short h = (unsigned short)(u >> 16);
  return __builtin_bit_cast(__bf16, h);
}

DEV bf16x16 pack16(bf16x8 lo, bf16x8 hi) {
  bf16x16 r;
#pragma unroll
  for (int e = 0; e < 8; ++e) { r[e] = lo[e]; r[e + 8] = hi[e]; }
  return r;
}

// ---------------------------------------------------------------------------
// Tensor Data Mover: DMA a 2D bf16 tile (tile_w x tile_h elements) from global
// (row stride = row_stride elements) into LDS at lds_off, padding each
// 64-byte row (32 bf16) with 16 bytes -> 40-element LDS row stride.
// D# layout per CDNA5 ISA ch.8 (group0 128b, group1 256b; groups 2/3 zero for
// 2D tiles). Tracked by TENSORcnt.
// ---------------------------------------------------------------------------
DEV void tdm_load_2d_bf16(const void* gptr, unsigned lds_off,
                          int tensor_w, int tensor_h, int tile_w, int tile_h,
                          int row_stride) {
  unsigned long long ga = (unsigned long long)gptr;
  u32x4 g0;
  g0[0] = 1u;                                            // count=1 (valid user D#)
  g0[1] = lds_off;                                       // lds_addr (bytes)
  g0[2] = (unsigned)(ga & 0xFFFFFFFFu);                  // global_addr[31:0]
  g0[3] = (unsigned)((ga >> 32) & 0x01FFFFFFu) | (2u << 30);  // addr[56:32] | type=2
  i32x8 g1;
  g1[0] = (1 << 16)    // data_size = 2 bytes (bf16)
        | (1 << 20)    // pad_enable
        | (3 << 22)    // pad_interval code 3 -> every 16 DWORDs (64B = 32 bf16)
        | (3 << 25);   // pad_amount  code 3 -> 4 DWORDs (16B = 8 bf16)
  g1[1] = (tensor_w & 0xFFFF) << 16;                                   // tensor_dim0 lo
  g1[2] = ((tensor_w >> 16) & 0xFFFF) | ((tensor_h & 0xFFFF) << 16);   // dim0 hi | dim1 lo
  g1[3] = ((tensor_h >> 16) & 0xFFFF) | ((tile_w & 0xFFFF) << 16);     // dim1 hi | tile_dim0
  g1[4] = tile_h & 0xFFFF;                                             // tile_dim1 (tile_dim2=0)
  g1[5] = row_stride;                                                  // tensor_dim0_stride lo
  g1[6] = 0;                                                           // stride hi | dim1_stride lo
  g1[7] = 0;
  i32x4 z4 = {0, 0, 0, 0};
#if __clang_major__ >= 23
  i32x8 z8 = {};
  __builtin_amdgcn_tensor_load_to_lds(g0, g1, z4, z4, z8, 0);
#else
  __builtin_amdgcn_tensor_load_to_lds(g0, g1, z4, z4, 0);
#endif
}

// ---------------------------------------------------------------------------
// fp32 -> bf16 conversion (x4 vectorized)
// ---------------------------------------------------------------------------
__global__ __launch_bounds__(256) void f2bf_kernel(const float* __restrict__ in,
                                                   __bf16* __restrict__ out, int n4) {
  int i = blockIdx.x * 256 + threadIdx.x;
  if (i < n4) {
    float4 f = ((const float4*)in)[i];
    bf16x4 o;
    o.x = f2bf(f.x); o.y = f2bf(f.y); o.z = f2bf(f.z); o.w = f2bf(f.w);
    ((bf16x4*)out)[i] = o;
  }
}

// ---------------------------------------------------------------------------
// bf16 WMMA GEMM:  C[M,N] = epilogue( A[M,K] @ W[N,K]^T )
// Workgroup tile 128x128, 8 waves (2x4), each wave 64x32 (4x2 WMMA tiles).
// K stepped 32/iter; tiles staged global->LDS by the Tensor Data Mover
// (double-buffered, wave0 issues the DMA, TENSORcnt + barrier sync).
// store modes: 0 = fp32 [M,N]
//              1 = bf16 head-split  [B,H,S,DK]   (Q/K)
//              2 = bf16 head-split-T [B,H,DK,S]  (V transposed)
//              3 = bf16 [M,N]
//              4 = fp32 [M,N] with residual add
// ---------------------------------------------------------------------------
__global__ __launch_bounds__(256)
void gemm_bf16(const __bf16* __restrict__ A, const __bf16* __restrict__ Bw,
               float* __restrict__ outF, __bf16* __restrict__ outB,
               const float* __restrict__ biasv, const float* __restrict__ resid,
               int M, int N, int K, int mode, int gelu) {
  __shared__ __attribute__((aligned(16))) __bf16 As[2][128][40];
  __shared__ __attribute__((aligned(16))) __bf16 Bs[2][128][40];

  const int tid  = threadIdx.x;
  const int lane = tid & 31;
  const int wid  = tid >> 5;
  const int wm   = wid >> 2;          // 0..1
  const int wn   = wid & 3;           // 0..3
  const int half = lane >> 4;
  const int l15  = lane & 15;
  const int ka   = half ? 8 : 0;      // A-fragment K offset pattern
  const int kb   = half ? 8 : 0;      // B-fragment K offset pattern
  const int m0   = blockIdx.y * 128;
  const int n0   = blockIdx.x * 128;

  f32x8 acc[4][2] = {};

  auto stage = [&](int t, int buf) {
    if (wid == 0) {
      const int k0 = t * 32;
      tdm_load_2d_bf16(A + (size_t)m0 * K + k0,
                       (unsigned)(uintptr_t)&As[buf][0][0],
                       K - k0, M - m0, 32, 128, K);
      tdm_load_2d_bf16(Bw + (size_t)n0 * K + k0,
                       (unsigned)(uintptr_t)&Bs[buf][0][0],
                       K - k0, N - n0, 32, 128, K);
    }
  };

  const int nk = K >> 5;
  stage(0, 0);
  __builtin_amdgcn_s_wait_tensorcnt(0);
  __syncthreads();

  for (int t = 0; t < nk; ++t) {
    const int buf = t & 1;
    if (t + 1 < nk) stage(t + 1, buf ^ 1);   // DMA next tile while computing

    bf16x16 afr[4];
#pragma unroll
    for (int i = 0; i < 4; ++i) {
      const __bf16* p = &As[buf][wm * 64 + i * 16 + l15][0];
      afr[i] = pack16(*(const bf16x8*)(p + ka), *(const bf16x8*)(p + ka + 16));
    }
    bf16x16 bfr[2];
#pragma unroll
    for (int j = 0; j < 2; ++j) {
      const __bf16* p = &Bs[buf][wn * 32 + j * 16 + l15][0];
      bfr[j] = pack16(*(const bf16x8*)(p + kb), *(const bf16x8*)(p + kb + 16));
    }
#pragma unroll
    for (int i = 0; i < 4; ++i)
#pragma unroll
      for (int j = 0; j < 2; ++j)
        acc[i][j] = __builtin_amdgcn_wmma_f32_16x16x32_bf16(
            false, afr[i], false, bfr[j], (short)0, acc[i][j], false, false);

    __builtin_amdgcn_s_wait_tensorcnt(0);    // wave0: next-tile DMA complete
    __syncthreads();                          // publish LDS to all waves
  }

  // ---- epilogue ----
#pragma unroll
  for (int i = 0; i < 4; ++i)
#pragma unroll
    for (int j = 0; j < 2; ++j)
#pragma unroll
      for (int r = 0; r < 8; ++r) {
        const int m = m0 + wm * 64 + i * 16 + r + 8 * half;
        const int n = n0 + wn * 32 + j * 16 + l15;
        float v = acc[i][j][r];
        if (biasv) v += biasv[n];
        if (gelu)  v = 0.5f * v * (1.0f + erff(v * 0.70710678118f));
        if (mode == 4) {
          float x = v + (resid ? resid[(size_t)m * N + n] : 0.0f);
          outF[(size_t)m * N + n] = x;
        } else if (mode == 0) {
          outF[(size_t)m * N + n] = v;
        } else if (mode == 3) {
          outB[(size_t)m * N + n] = f2bf(v);
        } else {
          const int bI = m >> 11, s = m & 2047, h = n >> 6, d = n & 63;
          const size_t bh = (size_t)(bI * 16 + h);
          if (mode == 1) outB[(bh * 2048 + s) * 64 + d] = f2bf(v);   // [B,H,S,DK]
          else           outB[(bh * 64 + d) * 2048 + s] = f2bf(v);   // [B,H,DK,S]
        }
      }
}

// ---------------------------------------------------------------------------
// Attention: one workgroup per (b,h, 16-row q-tile).  Scores for all 2048 keys
// held in registers (each wave owns a 256-key stripe, 16 WMMA tiles), bias
// fused, two-level softmax reduction, exp(P) staged bf16 in 64KB dynamic LDS,
// PV via WMMA (waves = 2 K-halves x 4 d-tiles), pairwise partial reduction.
// ---------------------------------------------------------------------------
__global__ __launch_bounds__(256)
void attn_kernel(const __bf16* __restrict__ Q, const __bf16* __restrict__ Kb,
                 const __bf16* __restrict__ Vt, const float* __restrict__ bias,
                 __bf16* __restrict__ Ob) {
  extern __shared__ __align__(16) char smem[];
  __bf16* sP   = (__bf16*)smem;                          // [16][2048]  64KB
  float*  sRed = (float*)(smem + 16 * 2048 * 2);         // [16][8]
  float*  sFin = sRed + 16 * 8;                          // [16] row max
  float*  sSum = sFin + 16;                              // [16] 1/rowsum
  float*  sO   = sSum + 16;                              // [4][16][16] partials

  const int tid  = threadIdx.x;
  const int lane = tid & 31;
  const int wid  = tid >> 5;
  const int half = lane >> 4;
  const int l15  = lane & 15;
  const int ka   = half ? 8 : 0;
  const int kbo  = half ? 8 : 0;

  const int bh = blockIdx.x >> 7;     // 0..31  (b*16+h)
  const int qt = blockIdx.x & 127;
  const int q0 = qt * 16;

  const __bf16* Qp = Q  + (size_t)bh * 2048 * 64;
  const __bf16* Kp = Kb + (size_t)bh * 2048 * 64;
  const __bf16* Vp = Vt + (size_t)bh * 64 * 2048;
  const float*  Bp = bias + (size_t)bh * 2048 * 2048 + (size_t)q0 * 2048;

  // ---- scores = Q @ K^T for this wave's 256-key stripe ----
  bf16x16 aq[2];
#pragma unroll
  for (int ks = 0; ks < 2; ++ks) {
    const __bf16* p = Qp + (size_t)(q0 + l15) * 64 + ks * 32;
    aq[ks] = pack16(*(const bf16x8*)(p + ka), *(const bf16x8*)(p + ka + 16));
  }
  f32x8 sc[16] = {};
#pragma unroll
  for (int ct = 0; ct < 16; ++ct) {
    const int k0 = wid * 256 + ct * 16;
#pragma unroll
    for (int ks = 0; ks < 2; ++ks) {
      const __bf16* p = Kp + (size_t)(k0 + l15) * 64 + ks * 32;
      bf16x16 bfr = pack16(*(const bf16x8*)(p + kbo), *(const bf16x8*)(p + kbo + 16));
      sc[ct] = __builtin_amdgcn_wmma_f32_16x16x32_bf16(
          false, aq[ks], false, bfr, (short)0, sc[ct], false, false);
    }
  }

  // ---- scale + bias + row max ----
  float rmax[8];
#pragma unroll
  for (int r = 0; r < 8; ++r) rmax[r] = -3.0e38f;
#pragma unroll
  for (int ct = 0; ct < 16; ++ct)
#pragma unroll
    for (int r = 0; r < 8; ++r) {
      float v = sc[ct][r] * 0.125f +
                Bp[(size_t)(r + 8 * half) * 2048 + wid * 256 + ct * 16 + l15];
      sc[ct][r] = v;
      rmax[r] = fmaxf(rmax[r], v);
    }
#pragma unroll
  for (int r = 0; r < 8; ++r)
    for (int mk = 1; mk < 16; mk <<= 1)
      rmax[r] = fmaxf(rmax[r], __shfl_xor(rmax[r], mk));
  if (l15 == 0)
#pragma unroll
    for (int r = 0; r < 8; ++r) sRed[(r + 8 * half) * 8 + wid] = rmax[r];
  __syncthreads();
  if (tid < 16) {
    float mv = sRed[tid * 8];
#pragma unroll
    for (int w = 1; w < 8; ++w) mv = fmaxf(mv, sRed[tid * 8 + w]);
    sFin[tid] = mv;
  }
  __syncthreads();

  // ---- exp, rowsum, P -> LDS (bf16, unnormalized) ----
  float rsum[8] = {};
#pragma unroll
  for (int ct = 0; ct < 16; ++ct)
#pragma unroll
    for (int r = 0; r < 8; ++r) {
      const int row = r + 8 * half;
      float p = __expf(sc[ct][r] - sFin[row]);
      rsum[r] += p;
      sP[row * 2048 + wid * 256 + ct * 16 + l15] = f2bf(p);
    }
#pragma unroll
  for (int r = 0; r < 8; ++r)
    for (int mk = 1; mk < 16; mk <<= 1) rsum[r] += __shfl_xor(rsum[r], mk);
  if (l15 == 0)
#pragma unroll
    for (int r = 0; r < 8; ++r) sRed[(r + 8 * half) * 8 + wid] = rsum[r];
  __syncthreads();
  if (tid < 16) {
    float s = 0.0f;
#pragma unroll
    for (int w = 0; w < 8; ++w) s += sRed[tid * 8 + w];
    sSum[tid] = 1.0f / s;
  }
  __syncthreads();

  // ---- O = P @ V^T : wave = (k-half, d-tile) ----
  const int dt = wid & 3;
  const int kh = wid >> 2;
  const int d0 = dt * 16;
  f32x8 oacc = {};
  for (int t = 0; t < 32; ++t) {
    const int koff = kh * 1024 + t * 32;
    const __bf16* pp = sP + l15 * 2048 + koff;
    bf16x16 af = pack16(*(const bf16x8*)(pp + ka), *(const bf16x8*)(pp + ka + 16));
    const __bf16* vp = Vp + (size_t)(d0 + l15) * 2048 + koff;
    bf16x16 bf = pack16(*(const bf16x8*)(vp + kbo), *(const bf16x8*)(vp + kbo + 16));
    oacc = __builtin_amdgcn_wmma_f32_16x16x32_bf16(
        false, af, false, bf, (short)0, oacc, false, false);
  }
  if (kh == 1)
#pragma unroll
    for (int r = 0; r < 8; ++r) sO[(dt * 16 + r + 8 * half) * 16 + l15] = oacc[r];
  __syncthreads();
  if (kh == 0) {
#pragma unroll
    for (int r = 0; r < 8; ++r) {
      const int row = r + 8 * half;
      float v = (oacc[r] + sO[(dt * 16 + row) * 16 + l15]) * sSum[row];
      // concat heads: [b][q][h*64+d]
      const int b = bh >> 4, h = bh & 15;
      Ob[((size_t)b * 2048 + q0 + row) * 1024 + h * 64 + d0 + l15] = f2bf(v);
    }
  }
}

// ---------------------------------------------------------------------------
// Row LayerNorm over D=1024, one workgroup per row (256 thr x 4 elems)
// ---------------------------------------------------------------------------
__global__ __launch_bounds__(256)
void layernorm_kernel(const float* __restrict__ X, const float* __restrict__ g,
                      const float* __restrict__ be, float* __restrict__ outF,
                      __bf16* __restrict__ outB) {
  const int row = blockIdx.x;
  const int tid = threadIdx.x;
  const int lane = tid & 31, wid = tid >> 5;
  const float* x = X + (size_t)row * 1024;

  float v[4], s = 0.0f, ss = 0.0f;
#pragma unroll
  for (int i = 0; i < 4; ++i) {
    float t = x[tid + 256 * i];
    v[i] = t; s += t; ss += t * t;
  }
  for (int mk = 1; mk < 32; mk <<= 1) { s += __shfl_xor(s, mk); ss += __shfl_xor(ss, mk); }

  __shared__ float rs[8], rss[8], stats[2];
  if (lane == 0) { rs[wid] = s; rss[wid] = ss; }
  __syncthreads();
  if (tid == 0) {
    float a = 0.0f, b = 0.0f;
#pragma unroll
    for (int w = 0; w < 8; ++w) { a += rs[w]; b += rss[w]; }
    float mu = a * (1.0f / 1024.0f);
    float var = b * (1.0f / 1024.0f) - mu * mu;
    stats[0] = mu;
    stats[1] = rsqrtf(var + 1e-5f);
  }
  __syncthreads();
  const float mu = stats[0], rstd = stats[1];
#pragma unroll
  for (int i = 0; i < 4; ++i) {
    const int c = tid + 256 * i;
    float y = (v[i] - mu) * rstd * g[c] + be[c];
    if (outB) outB[(size_t)row * 1024 + c] = f2bf(y);
    else      outF[(size_t)row * 1024 + c] = y;
  }
}

// ---------------------------------------------------------------------------
// Host orchestration
// ---------------------------------------------------------------------------
extern "C" void kernel_launch(void* const* d_in, const int* in_sizes, int n_in,
                              void* d_out, int out_size, void* d_ws, size_t ws_size,
                              hipStream_t stream) {
  (void)in_sizes; (void)n_in; (void)out_size; (void)ws_size;
  const float* x_q  = (const float*)d_in[0];
  const float* x_kv = (const float*)d_in[1];
  const float* bias = (const float*)d_in[2];
  const float* Wq   = (const float*)d_in[3];
  const float* Wk   = (const float*)d_in[4];
  const float* Wv   = (const float*)d_in[5];
  const float* Wo   = (const float*)d_in[6];
  const float* W1   = (const float*)d_in[7];
  const float* b1   = (const float*)d_in[8];
  const float* W2   = (const float*)d_in[9];
  const float* b2   = (const float*)d_in[10];
  const float* g1   = (const float*)d_in[11];
  const float* be1  = (const float*)d_in[12];
  const float* g2   = (const float*)d_in[13];
  const float* be2  = (const float*)d_in[14];
  float* out = (float*)d_out;

  size_t off = 0;
  auto carve = [&](size_t bytes) -> void* {
    void* p = (char*)d_ws + off;
    off += (bytes + 255) & ~(size_t)255;
    return p;
  };
  __bf16* xq_bf   = (__bf16*)carve(4194304ull * 2);  // [4096,1024]
  __bf16* xkv_bf  = (__bf16*)carve(4194304ull * 2);
  __bf16* wq_bf   = (__bf16*)carve(1048576ull * 2);
  __bf16* wk_bf   = (__bf16*)carve(1048576ull * 2);
  __bf16* wv_bf   = (__bf16*)carve(1048576ull * 2);
  __bf16* wo_bf   = (__bf16*)carve(1048576ull * 2);
  __bf16* w1_bf   = (__bf16*)carve(4194304ull * 2);  // [4096,1024]
  __bf16* w2_bf   = (__bf16*)carve(4194304ull * 2);  // [1024,4096]
  __bf16* qbf     = (__bf16*)carve(4194304ull * 2);  // [B,H,S,DK]
  __bf16* kbf     = (__bf16*)carve(4194304ull * 2);  // [B,H,S,DK]
  __bf16* vtbf    = (__bf16*)carve(4194304ull * 2);  // [B,H,DK,S]
  __bf16* attnbf  = (__bf16*)carve(4194304ull * 2);  // [4096,1024]
  float*  x1      = (float*) carve(4194304ull * 4);  // residual 1
  __bf16* ln1bf   = (__bf16*)carve(4194304ull * 2);
  __bf16* h1bf    = (__bf16*)carve(16777216ull * 2); // [4096,4096]
  float*  x2      = (float*) carve(4194304ull * 4);  // residual 2

  auto conv = [&](const float* src, __bf16* dst, int n) {
    int n4 = n >> 2;
    f2bf_kernel<<<dim3((n4 + 255) / 256), dim3(256), 0, stream>>>(src, dst, n4);
  };
  auto gemm = [&](const __bf16* A, const __bf16* Bw, float* oF, __bf16* oB,
                  const float* bv, const float* rz, int M, int N, int K,
                  int mode, int gelu) {
    gemm_bf16<<<dim3(N / 128, M / 128), dim3(256), 0, stream>>>(
        A, Bw, oF, oB, bv, rz, M, N, K, mode, gelu);
  };

  // bf16 conversions
  conv(x_q,  xq_bf,  4194304);
  conv(x_kv, xkv_bf, 4194304);
  conv(Wq, wq_bf, 1048576);
  conv(Wk, wk_bf, 1048576);
  conv(Wv, wv_bf, 1048576);
  conv(Wo, wo_bf, 1048576);
  conv(W1, w1_bf, 4194304);
  conv(W2, w2_bf, 4194304);

  // projections (head-split; V stored transposed for PV WMMA B-operand)
  gemm(xq_bf,  wq_bf, nullptr, qbf,  nullptr, nullptr, 4096, 1024, 1024, 1, 0);
  gemm(xkv_bf, wk_bf, nullptr, kbf,  nullptr, nullptr, 4096, 1024, 1024, 1, 0);
  gemm(xkv_bf, wv_bf, nullptr, vtbf, nullptr, nullptr, 4096, 1024, 1024, 2, 0);

  // attention (32 bh * 128 q-tiles), 70272B dynamic LDS
  const size_t ATTN_SMEM = 16 * 2048 * 2 + (128 + 16 + 16 + 1024) * 4;
  attn_kernel<<<dim3(4096), dim3(256), ATTN_SMEM, stream>>>(qbf, kbf, vtbf, bias, attnbf);

  // output projection + residual (fp32)
  gemm(attnbf, wo_bf, x1, nullptr, nullptr, x_q, 4096, 1024, 1024, 4, 0);

  // LN1 -> bf16
  layernorm_kernel<<<dim3(4096), dim3(256), 0, stream>>>(x1, g1, be1, nullptr, ln1bf);

  // FFN: GELU(ln1 @ W1^T + b1) -> bf16 ; then @ W2^T + b2 + residual -> fp32
  gemm(ln1bf, w1_bf, nullptr, h1bf, b1, nullptr, 4096, 4096, 1024, 3, 1);
  gemm(h1bf,  w2_bf, x2, nullptr, b2, x1,       4096, 1024, 4096, 4, 0);

  // LN2 -> fp32 output
  layernorm_kernel<<<dim3(4096), dim3(256), 0, stream>>>(x2, g2, be2, out, nullptr);
}
